// Attention_42365557407867
// MI455X (gfx1250) — compile-verified, matrix operand order checked by reading, not run on hardware
//
#include <hip/hip_runtime.h>

// ---------------------------------------------------------------------------
// MHA layer for MI455X (gfx1250, wave32, WMMA).
// x:[2,2048,1024] f32. All GEMMs + attention via v_wmma_f32_16x16x32_f16.
// Compute-bound (~70 GFLOP vs <10us of HBM traffic at 23.3TB/s): everything
// runs through f16 WMMA with f32 accumulation; attention is flash-style
// (scores never hit HBM). Softmax row-max uses DPP16 reductions (pure VALU,
// co-issues with the matrix pipe); row-sums come from a P @ ones WMMA whose
// C-tile layout (VGPR r <-> row r) is exactly the replication l_r[] needs.
// The 1/sqrt(64)*log2(e) scale is folded into the loop-invariant Q fragment.
// ---------------------------------------------------------------------------

typedef __attribute__((ext_vector_type(16))) _Float16 v16h;
typedef __attribute__((ext_vector_type(8)))  float    v8f;

#define DIM   1024
#define SEQ   2048
#define HEADS 16
#define HD    64
#define BATCH 2
#define MROWS (BATCH * SEQ)   // 4096

// ---- WMMA helper -----------------------------------------------------------
__device__ __forceinline__ v8f wmma32(v16h a, v16h b, v8f c) {
  return __builtin_amdgcn_wmma_f32_16x16x32_f16(
      /*neg_a=*/false, a, /*neg_b=*/false, b,
      /*c_mod=*/(short)0, c, /*reuse_a=*/false, /*reuse_b=*/false);
}

// 16-bit A/B fragment loader (16x32 tile, row-major source, 16B chunks).
// lane<16 : row = lane,    K = {k0+0..7, k0+16..23}
// lane>=16: row = lane-16, K = {k0+8..15, k0+24..31}
__device__ __forceinline__ v16h load_frag(const _Float16* base, int ld,
                                          int row, int k0) {
  const int lane = threadIdx.x & 31;
  const _Float16* p =
      base + (size_t)(row + (lane & 15)) * ld + (k0 + ((lane >> 4) << 3));
  v16h f;
  ((uint4*)&f)[0] = *(const uint4*)(p);
  ((uint4*)&f)[1] = *(const uint4*)(p + 16);
  return f;
}

// ---- DPP16 max-reduction across a 16-lane group (no LDS, no waits) ---------
template <int CTRL>
__device__ __forceinline__ float dpp_max_step(float x) {
  int xi = __builtin_bit_cast(int, x);
  int yi = __builtin_amdgcn_update_dpp(xi, xi, CTRL, 0xf, 0xf, true);
  return fmaxf(x, __builtin_bit_cast(float, yi));
}
__device__ __forceinline__ float red16_max(float x) {
  x = dpp_max_step<0xB1>(x);    // quad_perm [1,0,3,2]  (xor 1)
  x = dpp_max_step<0x4E>(x);    // quad_perm [2,3,0,1]  (xor 2)
  x = dpp_max_step<0x141>(x);   // row_half_mirror      (crosses 4-boundary)
  x = dpp_max_step<0x140>(x);   // row_mirror           (crosses 8-boundary)
  return x;
}

// ---- elementwise f32 -> f16 ------------------------------------------------
__global__ void __launch_bounds__(256)
f32_to_f16(const float* __restrict__ in, _Float16* __restrict__ out, int n) {
  int i = blockIdx.x * blockDim.x + threadIdx.x;
  int stride = gridDim.x * blockDim.x;
  for (; i < n; i += stride) out[i] = (_Float16)in[i];
}

// ---- W [K,N] f32 -> Wt [N,K] f16 (LDS-tiled transpose) ---------------------
__global__ void __launch_bounds__(256)
w_transpose_f16(const float* __restrict__ W, _Float16* __restrict__ Wt) {
  __shared__ float t[32][33];
  const int bn = blockIdx.x * 32;       // N tile
  const int bk = blockIdx.y * 32;       // K tile
  const int tx = threadIdx.x & 31;
  const int ty = threadIdx.x >> 5;      // 0..7
#pragma unroll
  for (int i = 0; i < 32; i += 8)
    t[ty + i][tx] = W[(size_t)(bk + ty + i) * DIM + bn + tx];
  __syncthreads();
#pragma unroll
  for (int i = 0; i < 32; i += 8)
    Wt[(size_t)(bn + ty + i) * DIM + bk + tx] = (_Float16)t[tx][ty + i];
}

// ---- V [B,S,DIM] f16 -> Vt [B,H,HD,S] f16 (per-head transpose) -------------
__global__ void __launch_bounds__(256)
v_transpose(const _Float16* __restrict__ v, _Float16* __restrict__ vt) {
  __shared__ _Float16 t[64][72];
  const int b = blockIdx.z, h = blockIdx.y, s0 = blockIdx.x * 64;
  const int tx = threadIdx.x & 63;
  const int ty = threadIdx.x >> 6;      // 0..3
  const _Float16* src = v + ((size_t)b * SEQ + s0) * DIM + h * HD;
#pragma unroll
  for (int i = 0; i < 64; i += 4)
    t[ty + i][tx] = src[(size_t)(ty + i) * DIM + tx];
  __syncthreads();
  _Float16* dst = vt + ((size_t)(b * HEADS + h)) * HD * SEQ + s0;
#pragma unroll
  for (int i = 0; i < 64; i += 4)
    dst[(size_t)(ty + i) * SEQ + tx] = t[tx][ty + i];
}

// ---- WMMA GEMM: C[M,N] = A[M,K] * Bt[N,K]^T + bias, wave tile 32x64 --------
template <typename OutT>
__global__ void __launch_bounds__(256)
gemm_wmma(const _Float16* __restrict__ A, const _Float16* __restrict__ Bt,
          const float* __restrict__ bias, OutT* __restrict__ C,
          int M, int N, int K) {
  const int wave = threadIdx.x >> 5;
  const int lane = threadIdx.x & 31;
  const int wm = wave & 1, wn = wave >> 1;        // 2 x 4 wave grid
  const int m0 = blockIdx.x * 64 + wm * 32;
  const int n0 = blockIdx.y * 256 + wn * 64;

  v8f acc[2][4] = {};
  for (int k0 = 0; k0 < K; k0 += 32) {
    v16h a0 = load_frag(A, K, m0, k0);
    v16h a1 = load_frag(A, K, m0 + 16, k0);
    if (k0 + 64 < K) {  // global_prefetch_b8 toward the next K panel
      __builtin_prefetch(A + (size_t)(m0 + (lane & 15)) * K + k0 + 64, 0, 1);
      __builtin_prefetch(Bt + (size_t)(n0 + (lane & 15)) * K + k0 + 64, 0, 1);
    }
#pragma unroll
    for (int j = 0; j < 4; ++j) {
      v16h bf = load_frag(Bt, K, n0 + 16 * j, k0);
      acc[0][j] = wmma32(a0, bf, acc[0][j]);
      acc[1][j] = wmma32(a1, bf, acc[1][j]);
    }
  }

  const int c = lane & 15;
  const int rowoff = (lane >> 4) * 8;   // C tile: VGPR r -> row r (+8 hi half)
#pragma unroll
  for (int i = 0; i < 2; ++i)
#pragma unroll
    for (int j = 0; j < 4; ++j) {
      const int n = n0 + 16 * j + c;
      const float bb = bias[n];
      const int mb = m0 + 16 * i + rowoff;
#pragma unroll
      for (int r = 0; r < 8; ++r)
        C[(size_t)(mb + r) * N + n] = (OutT)(acc[i][j][r] + bb);
    }
}

// ---- Flash attention: one wave per 16 query rows, 64-wide KV blocks --------
__global__ void __launch_bounds__(128)
attn_fa(const _Float16* __restrict__ q, const _Float16* __restrict__ k,
        const _Float16* __restrict__ vt, _Float16* __restrict__ emb) {
  __shared__ alignas(16) _Float16 Plds[4][16 * 64];   // per-wave P tile (2KB)
  const int wave = threadIdx.x >> 5;
  const int lane = threadIdx.x & 31;
  const int b = blockIdx.z, h = blockIdx.y;
  const int m0 = (blockIdx.x * 4 + wave) * 16;

  const _Float16* Q  = q  + ((size_t)b * SEQ + m0) * DIM + h * HD;
  const _Float16* Kb = k  + (size_t)b * SEQ * DIM + h * HD;
  const _Float16* Vt = vt + ((size_t)(b * HEADS + h)) * HD * SEQ;
  _Float16* P = Plds[wave];

  // Q fragments are loop-invariant; fold 1/sqrt(64)*log2(e) into them so the
  // score WMMA result is already in the exp2 domain (no per-block scaling).
  const _Float16 SC = (_Float16)(0.125f * 1.44269504088896340736f);
  v16h aq0 = load_frag(Q, DIM, 0, 0);
  v16h aq1 = load_frag(Q, DIM, 0, 32);
#pragma unroll
  for (int i = 0; i < 16; ++i) { aq0[i] *= SC; aq1[i] *= SC; }

  v16h vone;
#pragma unroll
  for (int i = 0; i < 16; ++i) vone[i] = (_Float16)1.0f;

  v8f o[4] = {};
  float m_r[8], l_r[8], cf[8];
#pragma unroll
  for (int r = 0; r < 8; ++r) { m_r[r] = -3.0e38f; l_r[r] = 0.f; }

  const int c = lane & 15;
  const int rowoff = (lane >> 4) * 8;

  for (int n0 = 0; n0 < SEQ; n0 += 64) {
    // ---- S' = (Q*SC) @ K^T (four 16x16 tiles, 8 WMMAs) ----
    v8f s[4] = {};
#pragma unroll
    for (int j = 0; j < 4; ++j) {
      v16h bk0 = load_frag(Kb, DIM, n0 + 16 * j, 0);
      v16h bk1 = load_frag(Kb, DIM, n0 + 16 * j, 32);
      s[j] = wmma32(aq1, bk1, wmma32(aq0, bk0, s[j]));
    }

    // ---- online softmax: row max via DPP16, p = exp2(s' - m') ----
#pragma unroll
    for (int r = 0; r < 8; ++r) {
      float mx = fmaxf(fmaxf(s[0][r], s[1][r]), fmaxf(s[2][r], s[3][r]));
      mx = red16_max(mx);
      const float nm = fmaxf(m_r[r], mx);
      cf[r]  = exp2f(m_r[r] - nm);
      m_r[r] = nm;
#pragma unroll
      for (int j = 0; j < 4; ++j) s[j][r] = exp2f(s[j][r] - nm);
    }
#pragma unroll
    for (int dt = 0; dt < 4; ++dt)
#pragma unroll
      for (int r = 0; r < 8; ++r) o[dt][r] *= cf[r];

    // ---- C-layout P -> A-frag layout via per-wave LDS tile ----
#pragma unroll
    for (int j = 0; j < 4; ++j)
#pragma unroll
      for (int r = 0; r < 8; ++r)
        P[(rowoff + r) * 64 + j * 16 + c] = (_Float16)s[j][r];
    __syncthreads();
    const v16h ap0 = load_frag(P, 64, 0, 0);    // kv 0..31  (ds_load_b128)
    const v16h ap1 = load_frag(P, 64, 0, 32);   // kv 32..63

    // ---- row sums via P @ ones: C layout == per-row replication for l ----
    v8f rs = wmma32(ap1, vone, wmma32(ap0, vone, (v8f){}));
#pragma unroll
    for (int r = 0; r < 8; ++r) l_r[r] = l_r[r] * cf[r] + rs[r];

    // ---- O += P @ V (8 WMMAs, Vt rows contiguous along seq) ----
#pragma unroll
    for (int dt = 0; dt < 4; ++dt) {
      v16h bv0 = load_frag(Vt, SEQ, dt * 16, n0);
      v16h bv1 = load_frag(Vt, SEQ, dt * 16, n0 + 32);
      o[dt] = wmma32(ap1, bv1, wmma32(ap0, bv0, o[dt]));
    }
    __syncthreads();
  }

  // ---- normalize and store f16 head slice ----
  float inv[8];
#pragma unroll
  for (int r = 0; r < 8; ++r) inv[r] = 1.0f / l_r[r];
  _Float16* E = emb + ((size_t)b * SEQ + m0) * DIM + h * HD;
#pragma unroll
  for (int dt = 0; dt < 4; ++dt)
#pragma unroll
    for (int r = 0; r < 8; ++r)
      E[(size_t)(rowoff + r) * DIM + dt * 16 + c] =
          (_Float16)(o[dt][r] * inv[r]);
}

// ---------------------------------------------------------------------------
extern "C" void kernel_launch(void* const* d_in, const int* in_sizes, int n_in,
                              void* d_out, int out_size, void* d_ws,
                              size_t ws_size, hipStream_t stream) {
  const float* x  = (const float*)d_in[0];
  const float* Wq = (const float*)d_in[1]; const float* bq = (const float*)d_in[2];
  const float* Wk = (const float*)d_in[3]; const float* bk = (const float*)d_in[4];
  const float* Wv = (const float*)d_in[5]; const float* bv = (const float*)d_in[6];
  const float* Wo = (const float*)d_in[7]; const float* bo = (const float*)d_in[8];
  float* out = (float*)d_out;

  // workspace layout (f16 elements); total 28M halves = 56 MB
  const size_t M1 = (size_t)1024 * 1024;
  _Float16* ws  = (_Float16*)d_ws;
  _Float16* xh  = ws;              // 4M
  _Float16* WqT = ws + 4 * M1;     // 1M each
  _Float16* WkT = ws + 5 * M1;
  _Float16* WvT = ws + 6 * M1;
  _Float16* WoT = ws + 7 * M1;
  _Float16* qh  = ws + 8 * M1;     // 4M each
  _Float16* kh  = ws + 12 * M1;
  _Float16* vh  = ws + 16 * M1;
  _Float16* vth = ws + 20 * M1;
  _Float16* eh  = ws + 24 * M1;

  // 1) precision conversion + weight transposes
  f32_to_f16<<<1024, 256, 0, stream>>>(x, xh, MROWS * DIM);
  dim3 wtg(32, 32);
  w_transpose_f16<<<wtg, 256, 0, stream>>>(Wq, WqT);
  w_transpose_f16<<<wtg, 256, 0, stream>>>(Wk, WkT);
  w_transpose_f16<<<wtg, 256, 0, stream>>>(Wv, WvT);
  w_transpose_f16<<<wtg, 256, 0, stream>>>(Wo, WoT);

  // 2) QKV projections (f16 out, +bias)
  dim3 gg(MROWS / 64, DIM / 256);
  gemm_wmma<_Float16><<<gg, 256, 0, stream>>>(xh, WqT, bq, qh, MROWS, DIM, DIM);
  gemm_wmma<_Float16><<<gg, 256, 0, stream>>>(xh, WkT, bk, kh, MROWS, DIM, DIM);
  gemm_wmma<_Float16><<<gg, 256, 0, stream>>>(xh, WvT, bv, vh, MROWS, DIM, DIM);

  // 3) per-head V transpose for contiguous PV operand
  v_transpose<<<dim3(SEQ / 64, HEADS, BATCH), 256, 0, stream>>>(vh, vth);

  // 4) flash attention (one wave / 16 query rows, KV block = 64)
  attn_fa<<<dim3(SEQ / 16 / 4, HEADS, BATCH), 128, 0, stream>>>(qh, kh, vth, eh);

  // 5) output projection (f32 out, +bias)
  gemm_wmma<float><<<gg, 256, 0, stream>>>(eh, WoT, bo, out, MROWS, DIM, DIM);
}
